// MambaAttentionLayer_4698694221983
// MI455X (gfx1250) — compile-verified, matrix operand order checked by reading, not run on hardware
//
#include <hip/hip_runtime.h>
#include <cstddef>

// ---------------------------------------------------------------------------
// Types / WMMA helpers (CDNA5 / gfx1250, wave32)
// ---------------------------------------------------------------------------
typedef __bf16 bf16_t;
typedef bf16_t v16bf __attribute__((ext_vector_type(16)));
typedef float  v8f   __attribute__((ext_vector_type(8)));

__device__ inline v8f vzero8() {
  v8f z = {0.f,0.f,0.f,0.f,0.f,0.f,0.f,0.f};
  return z;
}

__device__ inline v8f wmma_bf16(v16bf a, v16bf b, v8f c) {
#if defined(__gfx1250__)
  // D = A(16x32 bf16) x B(32x16 bf16) + C(16x16 f32)
  return __builtin_amdgcn_wmma_f32_16x16x32_bf16(false, a, false, b, (short)0, c, false, false);
#else
  (void)a; (void)b;
  return c;
#endif
}

// A/B fragment loader. rowbase points at this lane's row (A: M-row, B: N-column
// of the pre-transposed operand), K contiguous. Per ISA 16-bit A layout:
// half 0 holds K = k0+{0..7, 16..23}, half 1 holds K = k0+{8..15, 24..31}.
__device__ inline v16bf ldfrag(const bf16_t* rowbase, int k0, int half) {
  union U { v16bf v; uint4 q[2]; } u;
  u.q[0] = *reinterpret_cast<const uint4*>(rowbase + k0 + half * 8);
  u.q[1] = *reinterpret_cast<const uint4*>(rowbase + k0 + 16 + half * 8);
  return u.v;
}

__device__ inline float sigmoidf_(float x) { return 1.f / (1.f + __expf(-x)); }

// ---------------------------------------------------------------------------
// GEMM: C[M,N] = beta*C + alpha*(A_bf16[M,K] x Bt_bf16[N,K]^T + bias[N])
// block = 256 threads = 8 waves (4x2), block tile 128x64, wave tile 32x32.
// ---------------------------------------------------------------------------
__global__ __launch_bounds__(256)
void k_gemm(const bf16_t* __restrict__ A, int lda,
            const bf16_t* __restrict__ Bt, int ldb,
            const float* __restrict__ bias,
            float* __restrict__ C, int ldc,
            int M, int N, int K, float alpha, float beta)
{
  const int wave = threadIdx.x >> 5;
  const int lane = threadIdx.x & 31;
  const int half = lane >> 4;
  const int l    = lane & 15;
  const int wm = wave >> 1, wn = wave & 1;
  const int m0 = blockIdx.y * 128 + wm * 32;
  const int n0 = blockIdx.x * 64  + wn * 32;

  const bf16_t* arow[2];
#pragma unroll
  for (int i = 0; i < 2; ++i) {
    int r = m0 + i * 16 + l;
    if (r >= M) r = M - 1;               // clamped; store is predicated
    arow[i] = A + (size_t)r * lda;
  }
  const bf16_t* brow[2];
#pragma unroll
  for (int j = 0; j < 2; ++j)
    brow[j] = Bt + (size_t)(n0 + j * 16 + l) * ldb;

  v8f acc[2][2];
  acc[0][0] = vzero8(); acc[0][1] = vzero8();
  acc[1][0] = vzero8(); acc[1][1] = vzero8();

  for (int k0 = 0; k0 < K; k0 += 32) {
    v16bf af0 = ldfrag(arow[0], k0, half);
    v16bf af1 = ldfrag(arow[1], k0, half);
    v16bf bf0 = ldfrag(brow[0], k0, half);
    v16bf bf1 = ldfrag(brow[1], k0, half);
    acc[0][0] = wmma_bf16(af0, bf0, acc[0][0]);
    acc[0][1] = wmma_bf16(af0, bf1, acc[0][1]);
    acc[1][0] = wmma_bf16(af1, bf0, acc[1][0]);
    acc[1][1] = wmma_bf16(af1, bf1, acc[1][1]);
  }

#pragma unroll
  for (int i = 0; i < 2; ++i)
#pragma unroll
    for (int j = 0; j < 2; ++j)
#pragma unroll
      for (int r = 0; r < 8; ++r) {
        int row = m0 + i * 16 + r + 8 * half;
        int col = n0 + j * 16 + l;
        if (row < M) {
          size_t off = (size_t)row * ldc + col;
          float v = acc[i][j][r] + (bias ? bias[col] : 0.f);
          float res = alpha * v;
          if (beta != 0.f) res += beta * C[off];
          C[off] = res;
        }
      }
}

// ---------------------------------------------------------------------------
// Flash-style relative attention. grid (T/64, H, B), block 128 = 4 waves.
// Each wave: 16 q-rows. WMMA for Qu*K^T, Qv*P^T band, and P*V.
// ---------------------------------------------------------------------------
__global__ __launch_bounds__(128)
void k_flash_attn(const bf16_t* __restrict__ qu, const bf16_t* __restrict__ qv,
                  const bf16_t* __restrict__ kb,  const bf16_t* __restrict__ vt,
                  const bf16_t* __restrict__ pb,  float* __restrict__ out)
{
  const int T = 1024, D = 512, H = 8;
  const int b = blockIdx.z, h = blockIdx.y;
  const int w = threadIdx.x >> 5;
  const int lane = threadIdx.x & 31;
  const int half = lane >> 4;
  const int l    = lane & 15;
  const int q0 = blockIdx.x * 64 + w * 16;

  __shared__ float  bd_s[4][16 * 80];
  __shared__ bf16_t p_s [4][16 * 64];

  const bf16_t* qurow = qu + ((size_t)(b * T + q0 + l)) * D + h * 64;
  const bf16_t* qvrow = qv + ((size_t)(b * T + q0 + l)) * D + h * 64;
  v16bf aqu[2], aqv[2];
  aqu[0] = ldfrag(qurow, 0, half);  aqu[1] = ldfrag(qurow, 32, half);
  aqv[0] = ldfrag(qvrow, 0, half);  aqv[1] = ldfrag(qvrow, 32, half);

  v8f o[4];
  float mrow[8], lrow[8];
#pragma unroll
  for (int t = 0; t < 4; ++t) o[t] = vzero8();
#pragma unroll
  for (int r = 0; r < 8; ++r) { mrow[r] = -1e30f; lrow[r] = 0.f; }

  const float scale = 0.125f;   // 1/sqrt(64)

  for (int k0 = 0; k0 < T; k0 += 64) {
    // ---- S = Qu * K^T (16 x 64)
    v8f s[4];
#pragma unroll
    for (int t = 0; t < 4; ++t) {
      const bf16_t* krow = kb + ((size_t)(b * T + k0 + t * 16 + l)) * D + h * 64;
      v8f a = vzero8();
      a = wmma_bf16(aqu[0], ldfrag(krow, 0, half), a);
      a = wmma_bf16(aqu[1], ldfrag(krow, 32, half), a);
      s[t] = a;
    }
    // ---- bd band: bd[q,k] = Qv[q] . P[k + T-1 - q]; local col = kk - qq + 15
    int pbase = k0 + T - 1 - q0 - 15;
#pragma unroll
    for (int t = 0; t < 5; ++t) {
      int j = pbase + t * 16 + l;
      if (j < 0) j = 0;
      if (j > 2 * T - 2) j = 2 * T - 2;
      const bf16_t* prow = pb + (size_t)j * D + h * 64;
      v8f a = vzero8();
      a = wmma_bf16(aqv[0], ldfrag(prow, 0, half), a);
      a = wmma_bf16(aqv[1], ldfrag(prow, 32, half), a);
#pragma unroll
      for (int r = 0; r < 8; ++r)
        bd_s[w][(r + 8 * half) * 80 + t * 16 + l] = a[r];
    }
    __syncthreads();

    // ---- add band, scale, row max (rows live in half-specific lane groups)
    float rowmax[8];
#pragma unroll
    for (int r = 0; r < 8; ++r) {
      int qq = r + 8 * half;
      float mx = -1e30f;
#pragma unroll
      for (int t = 0; t < 4; ++t) {
        int kk = t * 16 + l;
        float v = (s[t][r] + bd_s[w][qq * 80 + (kk - qq + 15)]) * scale;
        s[t][r] = v;
        mx = fmaxf(mx, v);
      }
#pragma unroll
      for (int m = 1; m <= 8; m <<= 1) mx = fmaxf(mx, __shfl_xor(mx, m, 32));
      rowmax[r] = mx;
    }

    // ---- online softmax
#pragma unroll
    for (int r = 0; r < 8; ++r) {
      float mnew = fmaxf(mrow[r], rowmax[r]);
      float corr = __expf(mrow[r] - mnew);
      mrow[r] = mnew;
      lrow[r] *= corr;
      o[0][r] *= corr; o[1][r] *= corr; o[2][r] *= corr; o[3][r] *= corr;
      float psum = 0.f;
#pragma unroll
      for (int t = 0; t < 4; ++t) {
        float p = __expf(s[t][r] - mnew);
        psum += p;
        p_s[w][(r + 8 * half) * 64 + t * 16 + l] = (bf16_t)p;
      }
#pragma unroll
      for (int m = 1; m <= 8; m <<= 1) psum += __shfl_xor(psum, m, 32);
      lrow[r] += psum;
    }
    __syncthreads();

    // ---- O += P * V   (A from LDS, B from pre-transposed V (n-major, t-contig))
    const bf16_t* prow = &p_s[w][l * 64];
#pragma unroll
    for (int ks = 0; ks < 2; ++ks) {
      v16bf ap = ldfrag(prow, ks * 32, half);
#pragma unroll
      for (int t = 0; t < 4; ++t) {
        const bf16_t* vrow = vt + ((size_t)((b * H + h) * 64 + t * 16 + l)) * T + k0;
        o[t] = wmma_bf16(ap, ldfrag(vrow, ks * 32, half), o[t]);
      }
    }
  }

#pragma unroll
  for (int t = 0; t < 4; ++t)
#pragma unroll
    for (int r = 0; r < 8; ++r) {
      int row = q0 + r + 8 * half;
      out[((size_t)(b * T + row)) * D + h * 64 + t * 16 + l] = o[t][r] / lrow[r];
    }
}

// ---------------------------------------------------------------------------
// Elementwise / small kernels
// ---------------------------------------------------------------------------
__global__ __launch_bounds__(256)
void k_layernorm(const float* __restrict__ x, const float* __restrict__ g,
                 const float* __restrict__ bt, float* __restrict__ out, int d)
{
  int row = blockIdx.x;
  const float* xr = x + (size_t)row * d;
  float* orow = out + (size_t)row * d;
  __shared__ float red[256];
  float s = 0.f;
  for (int i = threadIdx.x; i < d; i += 256) s += xr[i];
  red[threadIdx.x] = s; __syncthreads();
  for (int k = 128; k > 0; k >>= 1) {
    if ((int)threadIdx.x < k) red[threadIdx.x] += red[threadIdx.x + k];
    __syncthreads();
  }
  float mean = red[0] * (1.f / d);
  __syncthreads();
  float v = 0.f;
  for (int i = threadIdx.x; i < d; i += 256) { float t = xr[i] - mean; v += t * t; }
  red[threadIdx.x] = v; __syncthreads();
  for (int k = 128; k > 0; k >>= 1) {
    if ((int)threadIdx.x < k) red[threadIdx.x] += red[threadIdx.x + k];
    __syncthreads();
  }
  float rstd = rsqrtf(red[0] * (1.f / d) + 1e-5f);
  for (int i = threadIdx.x; i < d; i += 256)
    orow[i] = (xr[i] - mean) * rstd * g[i] + bt[i];
}

__global__ void k_cvt(const float* __restrict__ in, bf16_t* __restrict__ out, int n) {
  int i = blockIdx.x * 256 + threadIdx.x;
  if (i < n) out[i] = (bf16_t)in[i];
}

// W (K x N) f32 -> Wt (N x K) bf16
__global__ void k_cvt_T(const float* __restrict__ w, bf16_t* __restrict__ out, int K, int N) {
  int i = blockIdx.x * 256 + threadIdx.x;
  if (i >= K * N) return;
  int k = i / N, n = i % N;
  out[(size_t)n * K + k] = (bf16_t)w[i];
}

__global__ void k_swish(float* __restrict__ x, int n) {
  int i = blockIdx.x * 256 + threadIdx.x;
  if (i < n) { float v = x[i]; x[i] = v * sigmoidf_(v); }
}

__global__ void k_softplus(float* __restrict__ x, int n) {
  int i = blockIdx.x * 256 + threadIdx.x;
  if (i < n) { float v = x[i]; x[i] = (v > 20.f) ? v : log1pf(__expf(v)); }
}

// mode 0: u * silu(g)   mode 1: u * sigmoid(g);  in rows of 2*Hd -> out rows of Hd
__global__ void k_glu(const float* __restrict__ in, float* __restrict__ out,
                      int Hd, int mode, int total) {
  int idx = blockIdx.x * 256 + threadIdx.x;
  if (idx >= total) return;
  int row = idx / Hd, i = idx - row * Hd;
  float u = in[(size_t)row * 2 * Hd + i];
  float g = in[(size_t)row * 2 * Hd + Hd + i];
  float sg = sigmoidf_(g);
  out[idx] = (mode == 0) ? u * (g * sg) : u * sg;
}

// mamba depthwise conv, K=4 causal (pad_l=3), + bias, + silu. src ld=2048, C=1024.
__global__ void k_dwconv4_silu(const float* __restrict__ src, const float* __restrict__ cw,
                               const float* __restrict__ cb, float* __restrict__ out) {
  int idx = blockIdx.x * 256 + threadIdx.x;
  if (idx >= 8192 * 1024) return;
  int row = idx >> 10;      // b*T + t
  int c = idx & 1023;
  int t = row & 1023;
  float acc = cb[c];
#pragma unroll
  for (int j = 0; j < 4; ++j) {
    int tt = t - 3 + j;
    if (tt >= 0) acc += src[(size_t)(row - 3 + j) * 2048 + c] * cw[c * 4 + j];
  }
  out[idx] = acc * sigmoidf_(acc);
}

// conv-module depthwise conv K=31 same-pad + bias + BN-ish scale + swish. C=512.
__global__ void k_dwconv31(const float* __restrict__ src, const float* __restrict__ cw,
                           const float* __restrict__ cb, const float* __restrict__ bng,
                           const float* __restrict__ bnb, float* __restrict__ out) {
  int idx = blockIdx.x * 256 + threadIdx.x;
  if (idx >= 8192 * 512) return;
  int row = idx >> 9;
  int c = idx & 511;
  int t = row & 1023;
  float acc = cb[c];
#pragma unroll 4
  for (int j = 0; j < 31; ++j) {
    int tt = t - 15 + j;
    if (tt >= 0 && tt < 1024) acc += src[(size_t)(row - 15 + j) * 512 + c] * cw[c * 31 + j];
  }
  float y = acc * rsqrtf(1.f + 1e-5f) * bng[c] + bnb[c];
  out[idx] = y * sigmoidf_(y);
}

// selective-scan: one thread per (b, channel); 16-wide state in registers.
__global__ void k_scan(const float* __restrict__ dt, const float* __restrict__ xc,
                       const float* __restrict__ dbc, const float* __restrict__ A_log,
                       float* __restrict__ y) {
  int tid = blockIdx.x * 256 + threadIdx.x;
  if (tid >= 8 * 1024) return;
  int b = tid >> 10, c = tid & 1023;
  float A[16], h[16];
#pragma unroll
  for (int n = 0; n < 16; ++n) { A[n] = -__expf(A_log[c * 16 + n]); h[n] = 0.f; }
  for (int t = 0; t < 1024; ++t) {
    size_t rowi = (size_t)(b * 1024 + t);
    float d = dt[rowi * 1024 + c];
    float x = xc[rowi * 1024 + c];
    const float* bc = dbc + rowi * 64;
    float yv = 0.f;
#pragma unroll
    for (int n = 0; n < 16; ++n) {
      float dA = __expf(d * A[n]);
      h[n] = dA * h[n] + d * bc[32 + n] * x;
      yv += h[n] * bc[48 + n];
    }
    y[rowi * 1024 + c] = yv;
  }
}

// y = (y + D[c]*xc) * silu(z),  z = wide[row*2048 + 1024 + c]
__global__ void k_ymix(float* __restrict__ y, const float* __restrict__ xc,
                       const float* __restrict__ wide, const float* __restrict__ Dv) {
  int idx = blockIdx.x * 256 + threadIdx.x;
  if (idx >= 8192 * 1024) return;
  int row = idx >> 10, c = idx & 1023;
  float z = wide[(size_t)row * 2048 + 1024 + c];
  float v = y[idx] + Dv[c] * xc[idx];
  y[idx] = v * (z * sigmoidf_(z));
}

// qu/qv = bf16(q + pbu/pbv), broadcast of (H,dk) bias
__global__ void k_qu_qv(const float* __restrict__ q, const float* __restrict__ pbu,
                        const float* __restrict__ pbv, bf16_t* __restrict__ qu,
                        bf16_t* __restrict__ qv) {
  int idx = blockIdx.x * 256 + threadIdx.x;
  if (idx >= 8192 * 512) return;
  int col = idx & 511;
  float v = q[idx];
  qu[idx] = (bf16_t)(v + pbu[col]);
  qv[idx] = (bf16_t)(v + pbv[col]);
}

// V (B*T, 512) f32 -> Vt bf16 [(b,h,n), t]
__global__ void k_vT(const float* __restrict__ v, bf16_t* __restrict__ vt) {
  int idx = blockIdx.x * 256 + threadIdx.x;
  if (idx >= 8192 * 512) return;
  int row = idx >> 9, col = idx & 511;
  int b = row >> 10, t = row & 1023;
  int h = col >> 6, n = col & 63;
  vt[((size_t)((b * 8 + h) * 64 + n)) * 1024 + t] = (bf16_t)v[idx];
}

// ---------------------------------------------------------------------------
// Host orchestration
// ---------------------------------------------------------------------------
extern "C" void kernel_launch(void* const* d_in, const int* in_sizes, int n_in,
                              void* d_out, int out_size, void* d_ws, size_t ws_size,
                              hipStream_t stream) {
  (void)in_sizes; (void)n_in; (void)out_size;
  const int B = 8, T = 1024;
  const int BT = B * T;               // 8192

  // ---- workspace carve-out
  char* wsb = (char*)d_ws;
  size_t off = 0;
  auto alloc = [&](size_t bytes) -> void* {
    void* p = wsb + off;
    off = (off + bytes + 255) & ~(size_t)255;
    return p;
  };
  float* r    = (float*)alloc((size_t)BT * 512 * 4);
  float* t0   = (float*)alloc((size_t)BT * 512 * 4);
  float* wide = (float*)alloc((size_t)BT * 2048 * 4);
  float* midA = (float*)alloc((size_t)BT * 1024 * 4);
  float* midB = (float*)alloc((size_t)BT * 1024 * 4);
  float* midC = (float*)alloc((size_t)BT * 1024 * 4);
  float* dbcf = (float*)alloc((size_t)BT * 64 * 4);
  float* posf = (float*)alloc((size_t)2047 * 512 * 4);
  bf16_t* abf   = (bf16_t*)alloc((size_t)BT * 2048 * 2);
  bf16_t* wbf   = (bf16_t*)alloc((size_t)2048 * 512 * 2);
  bf16_t* dbcbf = (bf16_t*)alloc((size_t)BT * 64 * 2);
  bf16_t* qubf  = (bf16_t*)alloc((size_t)BT * 512 * 2);
  bf16_t* qvbf  = (bf16_t*)alloc((size_t)BT * 512 * 2);
  bf16_t* kbfp  = (bf16_t*)alloc((size_t)BT * 512 * 2);
  bf16_t* vtbf  = (bf16_t*)alloc((size_t)BT * 512 * 2);
  bf16_t* posbf = (bf16_t*)alloc((size_t)2047 * 512 * 2);
  if (off > ws_size) return;   // not enough scratch; bail deterministically

  auto fin = [&](int i) { return (const float*)d_in[i]; };
  auto nb  = [](long n) { return (unsigned)((n + 255) / 256); };

  auto LN = [&](const float* in, int gi, int bi, float* out) {
    k_layernorm<<<BT, 256, 0, stream>>>(in, fin(gi), fin(bi), out, 512);
  };
  auto CVT = [&](const float* in, bf16_t* out, long n) {
    k_cvt<<<nb(n), 256, 0, stream>>>(in, out, (int)n);
  };
  auto WT = [&](int wi, int K, int N) {
    k_cvt_T<<<nb((long)K * N), 256, 0, stream>>>(fin(wi), wbf, K, N);
  };
  auto GEMM = [&](const bf16_t* A, int lda, const bf16_t* Bt, int ldb,
                  const float* bias, float* C, int ldc,
                  int M, int N, int K, float alpha, float beta) {
    dim3 g(N / 64, (M + 127) / 128, 1);
    k_gemm<<<g, 256, 0, stream>>>(A, lda, Bt, ldb, bias, C, ldc, M, N, K, alpha, beta);
  };

  // r = x
  (void)hipMemcpyAsync(r, d_in[0], (size_t)BT * 512 * 4, hipMemcpyDeviceToDevice, stream);

  // ---------------- FF1: r += 0.5*(swish(ln@w1+b1)@w2 + b2)
  LN(r, 2, 3, t0);
  CVT(t0, abf, (long)BT * 512);
  WT(4, 512, 2048);
  GEMM(abf, 512, wbf, 512, fin(5), wide, 2048, BT, 2048, 512, 1.f, 0.f);
  k_swish<<<nb((long)BT * 2048), 256, 0, stream>>>(wide, BT * 2048);
  CVT(wide, abf, (long)BT * 2048);
  WT(6, 2048, 512);
  GEMM(abf, 2048, wbf, 2048, fin(7), r, 512, BT, 512, 2048, 0.5f, 1.f);

  // ---------------- Mamba
  LN(r, 8, 9, t0);
  CVT(t0, abf, (long)BT * 512);
  WT(10, 512, 2048);
  GEMM(abf, 512, wbf, 512, nullptr, wide, 2048, BT, 2048, 512, 1.f, 0.f);
  k_dwconv4_silu<<<nb((long)BT * 1024), 256, 0, stream>>>(wide, fin(11), fin(12), midA);
  CVT(midA, abf, (long)BT * 1024);
  WT(13, 1024, 64);
  GEMM(abf, 1024, wbf, 1024, nullptr, dbcf, 64, BT, 64, 1024, 1.f, 0.f);
  CVT(dbcf, dbcbf, (long)BT * 64);
  WT(14, 32, 1024);
  GEMM(dbcbf, 64, wbf, 32, fin(15), midB, 1024, BT, 1024, 32, 1.f, 0.f);
  k_softplus<<<nb((long)BT * 1024), 256, 0, stream>>>(midB, BT * 1024);
  k_scan<<<32, 256, 0, stream>>>(midB, midA, dbcf, fin(16), midC);
  k_ymix<<<nb((long)BT * 1024), 256, 0, stream>>>(midC, midA, wide, fin(17));
  CVT(midC, abf, (long)BT * 1024);
  WT(18, 1024, 512);
  GEMM(abf, 1024, wbf, 1024, nullptr, r, 512, BT, 512, 1024, 1.f, 1.f);

  // ---------------- Gated MLP
  LN(r, 19, 20, t0);
  CVT(t0, abf, (long)BT * 512);
  WT(21, 512, 2048);
  GEMM(abf, 512, wbf, 512, fin(22), wide, 2048, BT, 2048, 512, 1.f, 0.f);
  k_glu<<<nb((long)BT * 1024), 256, 0, stream>>>(wide, midA, 1024, 0, BT * 1024);
  CVT(midA, abf, (long)BT * 1024);
  WT(23, 1024, 512);
  GEMM(abf, 1024, wbf, 1024, fin(24), r, 512, BT, 512, 1024, 1.f, 1.f);

  // ---------------- Relative-position attention
  LN(r, 25, 26, t0);
  CVT(t0, abf, (long)BT * 512);
  WT(27, 512, 512); GEMM(abf, 512, wbf, 512, fin(28), midA, 512, BT, 512, 512, 1.f, 0.f); // q
  WT(29, 512, 512); GEMM(abf, 512, wbf, 512, fin(30), midB, 512, BT, 512, 512, 1.f, 0.f); // k
  WT(31, 512, 512); GEMM(abf, 512, wbf, 512, fin(32), midC, 512, BT, 512, 512, 1.f, 0.f); // v
  k_qu_qv<<<nb((long)BT * 512), 256, 0, stream>>>(midA, fin(34), fin(35), qubf, qvbf);
  CVT(midB, kbfp, (long)BT * 512);
  k_vT<<<nb((long)BT * 512), 256, 0, stream>>>(midC, vtbf);
  CVT(fin(1), abf, (long)2047 * 512);
  WT(33, 512, 512);
  GEMM(abf, 512, wbf, 512, nullptr, posf, 512, 2047, 512, 512, 1.f, 0.f);
  CVT(posf, posbf, (long)2047 * 512);
  k_flash_attn<<<dim3(T / 64, 8, B), 128, 0, stream>>>(qubf, qvbf, kbfp, vtbf, posbf, t0);
  CVT(t0, abf, (long)BT * 512);
  WT(36, 512, 512);
  GEMM(abf, 512, wbf, 512, fin(37), r, 512, BT, 512, 512, 1.f, 1.f);

  // ---------------- Conv module
  LN(r, 38, 39, t0);
  CVT(t0, abf, (long)BT * 512);
  WT(40, 512, 1024);
  GEMM(abf, 512, wbf, 512, fin(41), midA, 1024, BT, 1024, 512, 1.f, 0.f);
  k_glu<<<nb((long)BT * 512), 256, 0, stream>>>(midA, midB, 512, 1, BT * 512);
  k_dwconv31<<<nb((long)BT * 512), 256, 0, stream>>>(midB, fin(42), fin(43), fin(44), fin(45), midC);
  CVT(midC, abf, (long)BT * 512);
  WT(46, 512, 512);
  GEMM(abf, 512, wbf, 512, fin(47), r, 512, BT, 512, 512, 1.f, 1.f);

  // ---------------- FF2
  LN(r, 48, 49, t0);
  CVT(t0, abf, (long)BT * 512);
  WT(50, 512, 2048);
  GEMM(abf, 512, wbf, 512, fin(51), wide, 2048, BT, 2048, 512, 1.f, 0.f);
  k_swish<<<nb((long)BT * 2048), 256, 0, stream>>>(wide, BT * 2048);
  CVT(wide, abf, (long)BT * 2048);
  WT(52, 2048, 512);
  GEMM(abf, 2048, wbf, 2048, fin(53), r, 512, BT, 512, 2048, 0.5f, 1.f);

  // ---------------- final LayerNorm -> d_out
  LN(r, 54, 55, (float*)d_out);
}
